// LiGhTIncoming_47124381172180
// MI455X (gfx1250) — matrix operand order repaired and checked by reading, not compiled, not run on hardware
//
#include <hip/hip_runtime.h>
#include <hip/hip_bf16.h>

// ---------------------------------------------------------------- constants
#define NN   4096
#define EE   65536
#define KK   16
#define DD   768
#define HH   12
#define LL   3
#define DT   64
#define DHH  64
#define PP   5
#define TD   2304   // 3*D
#define FD   3072   // 4*D

typedef __bf16 bf16;
typedef bf16  v16bf __attribute__((ext_vector_type(16)));
typedef float v8f   __attribute__((ext_vector_type(8)));

__device__ __forceinline__ float gelu_f(float x) {
    return 0.5f * x * (1.0f + erff(x * 0.70710678118654752440f));
}

// ---------------------------------------------------------------- fp32 -> bf16
__global__ void k_f32_to_bf16(const float* __restrict__ src, bf16* __restrict__ dst, int n) {
    int i = blockIdx.x * blockDim.x + threadIdx.x;
    if (i < n) dst[i] = (bf16)src[i];
}

// transpose+convert: src fp32 [K][Nc] -> dst bf16 [Nc][K]
__global__ void k_transpose_bf16(const float* __restrict__ src, bf16* __restrict__ dst,
                                 int K, int Nc) {
    int i = blockIdx.x * blockDim.x + threadIdx.x;
    if (i >= K * Nc) return;
    int k = i / Nc, n = i % Nc;
    dst[(size_t)n * K + k] = (bf16)src[i];
}

// ---------------------------------------------------------------- WMMA GEMM
// C[M][Nc] = act(A[M][K](bf16) * Wt[Nc][K]^T(bf16) + bias) (+ resid)
// Block = 128 threads (4 waves). Block tile = 64(M) x 64(N).
// Wave w computes rows m0+w*16..+15 over all 64 cols: 4 accumulators,
// one A fragment reused by 4 WMMAs per K-step.
// B tile (64 cols x 32 K = 4KB bf16) staged per-block into LDS with
// global_load_async_to_lds_b128 (ASYNCcnt), double-buffered.
// Requires: M % 64 == 0, Nc % 64 == 0, K % 32 == 0 (all callers satisfy).
__global__ __launch_bounds__(128)
void k_gemm_bf16(const bf16* __restrict__ A, const bf16* __restrict__ Wt,
                 const float* __restrict__ bias, const float* __restrict__ resid,
                 float* __restrict__ outF, bf16* __restrict__ outB,
                 int M, int Nc, int K, int do_gelu) {
    __shared__ alignas(16) unsigned char smem[2][64 * 64];   // [buf][col*64B k-bytes]

    const int tid  = threadIdx.x;
    const int lane = tid & 31;
    const int wave = tid >> 5;
    const int n0   = blockIdx.x * 64;
    const int m0   = blockIdx.y * 64 + wave * 16;

    const int lo16 = lane & 15;
    const int hi   = lane >> 4;
    const int ksel = hi * 8;        // A-fragment K offset (ISA 16-bit A 16x32 layout)

    // --- async staging addresses: 256 chunks of 16B; thread t covers chunks t, t+128
    const int c0 = tid >> 2;        // col 0..31
    const int q0 = tid & 3;         // 16B quarter within the 64B k-line
    const bf16* gB0 = Wt + (size_t)(n0 + c0)      * K + q0 * 8;
    const bf16* gB1 = Wt + (size_t)(n0 + c0 + 32) * K + q0 * 8;
    // LDS byte offsets (low 32 bits of generic shared address == LDS offset)
    const unsigned ldsB0[2] = {
        (unsigned)(size_t)&smem[0][c0 * 64 + q0 * 16],
        (unsigned)(size_t)&smem[1][c0 * 64 + q0 * 16] };
    const unsigned ldsB1[2] = {
        (unsigned)(size_t)&smem[0][(c0 + 32) * 64 + q0 * 16],
        (unsigned)(size_t)&smem[1][(c0 + 32) * 64 + q0 * 16] };

    const bf16* pa = A + (size_t)(m0 + lo16) * K;

    v8f acc[4];
#pragma unroll
    for (int j = 0; j < 4; ++j)
        acc[j] = (v8f){0.f, 0.f, 0.f, 0.f, 0.f, 0.f, 0.f, 0.f};

    const int nk = K >> 5;

    // prologue: stage k-step 0 into buffer 0
    asm volatile("global_load_async_to_lds_b128 %0, %1, off"
                 :: "v"(ldsB0[0]), "v"((const void*)(gB0)) : "memory");
    asm volatile("global_load_async_to_lds_b128 %0, %1, off"
                 :: "v"(ldsB1[0]), "v"((const void*)(gB1)) : "memory");

    for (int ks = 0; ks < nk; ++ks) {
        const int buf = ks & 1;
        if (ks + 1 < nk) {
            const bf16* g0 = gB0 + (ks + 1) * 32;
            const bf16* g1 = gB1 + (ks + 1) * 32;
            asm volatile("global_load_async_to_lds_b128 %0, %1, off"
                         :: "v"(ldsB0[buf ^ 1]), "v"((const void*)g0) : "memory");
            asm volatile("global_load_async_to_lds_b128 %0, %1, off"
                         :: "v"(ldsB1[buf ^ 1]), "v"((const void*)g1) : "memory");
            asm volatile("s_wait_asynccnt 0x2" ::: "memory");   // batch ks done
        } else {
            asm volatile("s_wait_asynccnt 0x0" ::: "memory");
        }
        __syncthreads();   // whole B tile for k-step ks visible in LDS

        const int k = ks * 32;
        union Frag { v16bf v; uint4 q[2]; } a, b;
        a.q[0] = *(const uint4*)(pa + k + ksel);        // K = ksel..ksel+7
        a.q[1] = *(const uint4*)(pa + k + ksel + 16);   // K = ksel+16..ksel+23
        if (ks + 1 < nk) __builtin_prefetch(pa + k + 32, 0, 0);  // global_prefetch_b8

#pragma unroll
        for (int j = 0; j < 4; ++j) {
            // B fragment: lane -> col j*16+(l&15); K = hi*8..+7 and 16+hi*8..+7
            const uint4* pb =
                (const uint4*)&smem[buf][(j * 16 + lo16) * 64 + hi * 16];
            b.q[0] = pb[0];
            b.q[1] = pb[2];
            acc[j] = __builtin_amdgcn_wmma_f32_16x16x32_bf16(
                         false, a.v, false, b.v, (short)0, acc[j], false, false);
        }
        __syncthreads();   // all waves done reading buf before it is re-staged
    }

    // epilogue: lane l -> col (l&15); VGPR r -> row m0 + r + (l>>4)*8
    const int mb = m0 + hi * 8;
#pragma unroll
    for (int j = 0; j < 4; ++j) {
        const int nc = n0 + j * 16 + lo16;
        const float bv = bias ? bias[nc] : 0.0f;
#pragma unroll
        for (int r = 0; r < 8; ++r) {
            float v = acc[j][r] + bv;
            if (do_gelu) v = gelu_f(v);
            size_t off = (size_t)(mb + r) * Nc + nc;
            if (resid) v += resid[off];
            if (outF)  outF[off] = v;
            if (outB)  outB[off] = (bf16)v;
        }
    }
}

// ---------------------------------------------------------------- dist-attn LUT
// rows 0..5: path_len_emb; row 6: virtual_path_emb; row 7: self_loop_emb
__global__ void k_dist_lut(const float* __restrict__ plen_emb,
                           const float* __restrict__ vemb,
                           const float* __restrict__ semb,
                           const float* __restrict__ W1, const float* __restrict__ b1,
                           const float* __restrict__ W2, const float* __restrict__ b2,
                           float* __restrict__ lut) {
    __shared__ float emb[DD];
    __shared__ float hid[DD];
    const int r = blockIdx.x;
    const float* e = (r < PP + 1) ? (plen_emb + (size_t)r * DD) : (r == PP + 1 ? vemb : semb);
    for (int i = threadIdx.x; i < DD; i += blockDim.x) emb[i] = e[i];
    __syncthreads();
    for (int c = threadIdx.x; c < DD; c += blockDim.x) {
        float acc = b1[c];
        for (int d = 0; d < DD; ++d) acc += emb[d] * W1[(size_t)d * DD + c];
        hid[c] = gelu_f(acc);
    }
    __syncthreads();
    for (int hh = threadIdx.x; hh < HH; hh += blockDim.x) {
        float acc = b2[hh];
        for (int c = 0; c < DD; ++c) acc += hid[c] * W2[(size_t)c * HH + hh];
        lut[r * HH + hh] = acc;
    }
}

// ---------------------------------------------------------------- path MLP + edge bias
__global__ void k_path_eb(const float* __restrict__ proj2,       // [P][N][DT]
                          const int* __restrict__ path,          // [E][P]
                          const unsigned char* __restrict__ vp,
                          const unsigned char* __restrict__ sl,
                          const unsigned char* __restrict__ mask_e,
                          const float* __restrict__ lut,         // [8][H]
                          const float* __restrict__ W1, const float* __restrict__ b1,
                          const float* __restrict__ W2, const float* __restrict__ b2,
                          float* __restrict__ eb, int E) {
    int e = blockIdx.x * blockDim.x + threadIdx.x;
    if (e >= E) return;

    float ph[DT];
#pragma unroll
    for (int j = 0; j < DT; ++j) ph[j] = 0.f;
    int cnt = 0, plen = 0;
    for (int i = 0; i < PP; ++i) {
        int p0 = path[e * PP + i];
        int p  = (p0 < -99) ? -1 : p0;
        if (p0 >= 0) plen++;
        if (p >= 0) {
            cnt++;
            const float* row = proj2 + ((size_t)i * NN + p) * DT;
            for (int j = 0; j < DT; ++j) ph[j] += row[j];
        }
    }
    float inv = 1.0f / fmaxf((float)cnt, 1.0f);
#pragma unroll
    for (int j = 0; j < DT; ++j) ph[j] *= inv;

    float oh[HH];
#pragma unroll
    for (int hh = 0; hh < HH; ++hh) oh[hh] = b2[hh];
    for (int c = 0; c < DT; ++c) {
        float acc = b1[c];
        for (int j = 0; j < DT; ++j) acc += ph[j] * W1[j * DT + c];
        float g = gelu_f(acc);
#pragma unroll
        for (int hh = 0; hh < HH; ++hh) oh[hh] += g * W2[c * HH + hh];
    }
    int idx = sl[e] ? 7 : (vp[e] ? 6 : plen);
    bool me = mask_e[e] != 0;
#pragma unroll
    for (int hh = 0; hh < HH; ++hh)
        eb[(size_t)e * HH + hh] = me ? (lut[idx * HH + hh] + oh[hh]) : 0.f;
}

// ---------------------------------------------------------------- LayerNorm -> bf16
__global__ void k_layernorm(const float* __restrict__ x, const float* __restrict__ g,
                            const float* __restrict__ b, bf16* __restrict__ out) {
    const int row = blockIdx.x;
    const float* xr = x + (size_t)row * DD;
    __shared__ float red[256];
    float s = 0.f;
    for (int i = threadIdx.x; i < DD; i += 256) s += xr[i];
    red[threadIdx.x] = s; __syncthreads();
    for (int o = 128; o > 0; o >>= 1) { if (threadIdx.x < o) red[threadIdx.x] += red[threadIdx.x + o]; __syncthreads(); }
    float mean = red[0] / (float)DD; __syncthreads();
    float v = 0.f;
    for (int i = threadIdx.x; i < DD; i += 256) { float d = xr[i] - mean; v += d * d; }
    red[threadIdx.x] = v; __syncthreads();
    for (int o = 128; o > 0; o >>= 1) { if (threadIdx.x < o) red[threadIdx.x] += red[threadIdx.x + o]; __syncthreads(); }
    float rstd = rsqrtf(red[0] / (float)DD + 1e-5f);
    for (int i = threadIdx.x; i < DD; i += 256)
        out[(size_t)row * DD + i] = (bf16)((xr[i] - mean) * rstd * g[i] + b[i]);
}

// ---------------------------------------------------------------- edge QK dots
__global__ void k_edge_dot(const float* __restrict__ qkv, const int* __restrict__ src,
                           const int* __restrict__ dst, const unsigned char* __restrict__ mask_e,
                           const float* __restrict__ eb, float* __restrict__ scores, int E) {
    int t = blockIdx.x * blockDim.x + threadIdx.x;
    if (t >= E * HH) return;
    int e = t / HH, hh = t - e * HH;
    bool me = mask_e[e] != 0;
    int se = me ? src[e] : 0;
    int de = me ? dst[e] : 0;
    const float* q = qkv + (size_t)se * TD + hh * DHH;          // q block
    const float* k = qkv + (size_t)de * TD + DD + hh * DHH;     // k block
    float dot = 0.f;
#pragma unroll
    for (int d = 0; d < DHH; ++d) dot += q[d] * k[d];
    dot *= 0.036084391824351615f;                               // 768^-0.5
    scores[t] = (me ? dot : 0.f) + eb[t];
}

// ---------------------------------------------------------------- softmax + aggregate
__global__ void k_attn_agg(const float* __restrict__ qkv, const float* __restrict__ scores,
                           const int* __restrict__ inc_idx, const unsigned char* __restrict__ inc_mask,
                           const unsigned char* __restrict__ mask_nodes, const int* __restrict__ src,
                           const unsigned char* __restrict__ mask_e, bf16* __restrict__ out) {
    int t = blockIdx.x * blockDim.x + threadIdx.x;
    if (t >= NN * HH) return;
    int n = t / HH, hh = t - n * HH;
    bool mn = mask_nodes[n] != 0;

    float s[KK]; int ei[KK]; float vm[KK];
    float mx = -3.0e38f;
#pragma unroll
    for (int k = 0; k < KK; ++k) {
        int ii = inc_idx[n * KK + k];
        int e  = ii > 0 ? ii : 0;
        ei[k] = e;
        bool m = (inc_mask[n * KK + k] != 0) && mn;
        float sc = m ? scores[(size_t)e * HH + hh] : -1.0e9f;
        s[k] = sc;
        mx = fmaxf(mx, sc);
        vm[k] = (mask_e[e] != 0) ? 1.f : 0.f;
    }
    float den = 0.f;
#pragma unroll
    for (int k = 0; k < KK; ++k) { s[k] = __expf(s[k] - mx); den += s[k]; }
    float inv = 1.0f / den;

    float o[DHH];
#pragma unroll
    for (int d = 0; d < DHH; ++d) o[d] = 0.f;
    for (int k = 0; k < KK; ++k) {
        float a = s[k] * inv * vm[k];
        int se = (mask_e[ei[k]] != 0) ? src[ei[k]] : 0;
        const float* v = qkv + (size_t)se * TD + 2 * DD + hh * DHH;
#pragma unroll
        for (int d = 0; d < DHH; ++d) o[d] += a * v[d];
    }
    bf16* op = out + ((size_t)n * HH + hh) * DHH;
#pragma unroll
    for (int d = 0; d < DHH; ++d) op[d] = (bf16)o[d];
}

// ---------------------------------------------------------------- host orchestration
static inline size_t alignup(size_t x) { return (x + 255) & ~(size_t)255; }

extern "C" void kernel_launch(void* const* d_in, const int* in_sizes, int n_in,
                              void* d_out, int out_size, void* d_ws, size_t ws_size,
                              hipStream_t stream) {
    const float* triplet_h = (const float*)d_in[0];
    const unsigned char* mask_nodes = (const unsigned char*)d_in[1];
    const int* src  = (const int*)d_in[2];
    const int* dst  = (const int*)d_in[3];
    const int* path = (const int*)d_in[4];
    const unsigned char* vp  = (const unsigned char*)d_in[5];
    const unsigned char* sl  = (const unsigned char*)d_in[6];
    const unsigned char* mask_edges = (const unsigned char*)d_in[7];
    const int* inc_idx = (const int*)d_in[8];
    const unsigned char* inc_mask = (const unsigned char*)d_in[9];
    const float* path_len_emb = (const float*)d_in[10];
    const float* virtual_path_emb = (const float*)d_in[11];
    const float* self_loop_emb = (const float*)d_in[12];
    const float* dist_W1 = (const float*)d_in[13];
    const float* dist_b1 = (const float*)d_in[14];
    const float* dist_W2 = (const float*)d_in[15];
    const float* dist_b2 = (const float*)d_in[16];
    const float* trip_Win  = (const float*)d_in[17];
    const float* trip_bin  = (const float*)d_in[18];
    const float* trip_Wout = (const float*)d_in[19];
    const float* trip_bout = (const float*)d_in[20];
    const float* pattn_W1 = (const float*)d_in[21];
    const float* pattn_b1 = (const float*)d_in[22];
    const float* pattn_W2 = (const float*)d_in[23];
    const float* pattn_b2 = (const float*)d_in[24];
    const float* ln1_g = (const float*)d_in[25];
    const float* ln1_b = (const float*)d_in[26];
    const float* qkv_W = (const float*)d_in[27];
    const float* qkv_b = (const float*)d_in[28];
    const float* res_ln_g = (const float*)d_in[29];
    const float* res_ln_b = (const float*)d_in[30];
    const float* res_inW = (const float*)d_in[31];
    const float* res_inb = (const float*)d_in[32];
    const float* ffn_W1 = (const float*)d_in[33];
    const float* ffn_b1 = (const float*)d_in[34];
    const float* ffn_W2 = (const float*)d_in[35];
    const float* ffn_b2 = (const float*)d_in[36];

    // ---- workspace bump allocator
    char* w = (char*)d_ws;
    auto alloc = [&](size_t bytes) { char* p = w; w += alignup(bytes); return p; };

    bf16*  th_bf   = (bf16*)alloc((size_t)NN * DD * 2);
    bf16*  qkvWt   = (bf16*)alloc((size_t)LL * TD * DD * 2);   // [l][2304][768]
    bf16*  resWt   = (bf16*)alloc((size_t)LL * DD * DD * 2);   // [l][768][768]
    bf16*  f1Wt    = (bf16*)alloc((size_t)LL * FD * DD * 2);   // [l][3072][768]
    bf16*  f2Wt    = (bf16*)alloc((size_t)LL * DD * FD * 2);   // [l][768][3072]
    bf16*  winT    = (bf16*)alloc((size_t)PP * DT * DD * 2);   // [p][64][768]
    bf16*  woutT   = (bf16*)alloc((size_t)PP * DT * DT * 2);   // [p][64][64]
    float* lut     = (float*)alloc(8 * HH * 4);
    bf16*  proj1_bf= (bf16*)alloc((size_t)NN * DT * 2);
    float* proj2   = (float*)alloc((size_t)PP * NN * DT * 4);
    float* eb      = (float*)alloc((size_t)EE * HH * 4);
    bf16*  x_bf    = (bf16*)alloc((size_t)NN * DD * 2);
    float* qkv_f   = (float*)alloc((size_t)NN * TD * 4);
    float* scores  = (float*)alloc((size_t)EE * HH * 4);
    bf16*  out_bf  = (bf16*)alloc((size_t)NN * DD * 2);
    float* x2      = (float*)alloc((size_t)NN * DD * 4);
    bf16*  y_bf    = (bf16*)alloc((size_t)NN * DD * 2);
    bf16*  mid_bf  = (bf16*)alloc((size_t)NN * FD * 2);
    float* h_buf   = (float*)alloc((size_t)NN * DD * 4);
    (void)ws_size; (void)n_in; (void)in_sizes; (void)out_size;

    const int TB = 256;
    auto cdiv = [](long long a, long long b) { return (unsigned)((a + b - 1) / b); };

    // 1) activations and weights -> bf16 (weights transposed to [Ncol][K])
    k_f32_to_bf16<<<cdiv((long long)NN * DD, TB), TB, 0, stream>>>(triplet_h, th_bf, NN * DD);
    for (int l = 0; l < LL; ++l) {
        k_transpose_bf16<<<cdiv((long long)DD * TD, TB), TB, 0, stream>>>(
            qkv_W + (size_t)l * DD * TD, qkvWt + (size_t)l * TD * DD, DD, TD);
        k_transpose_bf16<<<cdiv((long long)DD * DD, TB), TB, 0, stream>>>(
            res_inW + (size_t)l * DD * DD, resWt + (size_t)l * DD * DD, DD, DD);
        k_transpose_bf16<<<cdiv((long long)DD * FD, TB), TB, 0, stream>>>(
            ffn_W1 + (size_t)l * DD * FD, f1Wt + (size_t)l * FD * DD, DD, FD);
        k_transpose_bf16<<<cdiv((long long)FD * DD, TB), TB, 0, stream>>>(
            ffn_W2 + (size_t)l * FD * DD, f2Wt + (size_t)l * DD * FD, FD, DD);
    }
    for (int p = 0; p < PP; ++p) {
        k_transpose_bf16<<<cdiv((long long)DD * DT, TB), TB, 0, stream>>>(
            trip_Win + (size_t)p * DD * DT, winT + (size_t)p * DT * DD, DD, DT);
        k_transpose_bf16<<<cdiv((long long)DT * DT, TB), TB, 0, stream>>>(
            trip_Wout + (size_t)p * DT * DT, woutT + (size_t)p * DT * DT, DT, DT);
    }

    // 2) dist-attn LUT (8 distinct embeddings)
    k_dist_lut<<<8, 256, 0, stream>>>(path_len_emb, virtual_path_emb, self_loop_emb,
                                      dist_W1, dist_b1, dist_W2, dist_b2, lut);

    // 3) triplet projections via WMMA: proj2[p] = gelu(th@Win[p]+bin)@Wout[p]+bout
    for (int p = 0; p < PP; ++p) {
        dim3 g1(DT / 64, NN / 64);
        k_gemm_bf16<<<g1, 128, 0, stream>>>(th_bf, winT + (size_t)p * DT * DD,
            trip_bin + p * DT, nullptr, nullptr, proj1_bf, NN, DT, DD, 1);
        k_gemm_bf16<<<g1, 128, 0, stream>>>(proj1_bf, woutT + (size_t)p * DT * DT,
            trip_bout + p * DT, nullptr, proj2 + (size_t)p * NN * DT, nullptr, NN, DT, DT, 0);
    }

    // 4) fused path-MLP + dist LUT + edge mask -> edge_bias
    k_path_eb<<<cdiv(EE, TB), TB, 0, stream>>>(proj2, path, vp, sl, mask_edges, lut,
                                               pattn_W1, pattn_b1, pattn_W2, pattn_b2, eb, EE);

    // 5) h = triplet_h
    hipMemcpyAsync(h_buf, triplet_h, (size_t)NN * DD * 4, hipMemcpyDeviceToDevice, stream);

    // 6) transformer layers
    for (int l = 0; l < LL; ++l) {
        k_layernorm<<<NN, 256, 0, stream>>>(h_buf, ln1_g + (size_t)l * DD, ln1_b + (size_t)l * DD, x_bf);

        dim3 gq(TD / 64, NN / 64);
        k_gemm_bf16<<<gq, 128, 0, stream>>>(x_bf, qkvWt + (size_t)l * TD * DD,
            qkv_b + (size_t)l * TD, nullptr, qkv_f, nullptr, NN, TD, DD, 0);

        k_edge_dot<<<cdiv((long long)EE * HH, TB), TB, 0, stream>>>(
            qkv_f, src, dst, mask_edges, eb, scores, EE);

        k_attn_agg<<<cdiv((long long)NN * HH, TB), TB, 0, stream>>>(
            qkv_f, scores, inc_idx, inc_mask, mask_nodes, src, mask_edges, out_bf);

        dim3 gr(DD / 64, NN / 64);
        k_gemm_bf16<<<gr, 128, 0, stream>>>(out_bf, resWt + (size_t)l * DD * DD,
            res_inb + (size_t)l * DD, h_buf, x2, nullptr, NN, DD, DD, 0);

        k_layernorm<<<NN, 256, 0, stream>>>(x2, res_ln_g + (size_t)l * DD, res_ln_b + (size_t)l * DD, y_bf);

        dim3 gf1(FD / 64, NN / 64);
        k_gemm_bf16<<<gf1, 128, 0, stream>>>(y_bf, f1Wt + (size_t)l * FD * DD,
            ffn_b1 + (size_t)l * FD, nullptr, nullptr, mid_bf, NN, FD, DD, 1);

        float* hout = (l == LL - 1) ? (float*)d_out : h_buf;
        dim3 gf2(DD / 64, NN / 64);
        k_gemm_bf16<<<gf2, 128, 0, stream>>>(mid_bf, f2Wt + (size_t)l * DD * FD,
            ffn_b2 + (size_t)l * DD, x2, hout, nullptr, NN, DD, FD, 0);
    }
}
